// MLP_22565758173770
// MI455X (gfx1250) — compile-verified, matrix operand order checked by reading, not run on hardware
//
#include <hip/hip_runtime.h>
#include <hip/hip_bf16.h>
#include <stdint.h>

// ---------------------------------------------------------------------------
// NF4-quantized SwiGLU MLP for MI455X (gfx1250, wave32, WMMA).
//   out = ( silu(x @ Wg^T) * (x @ Wu^T) ) @ Wd^T
// HBM-bound: 3 x 11008 x 4096 int32 codes = 541 MB read exactly once; dequant
// is fused into the GEMM. Double-buffered LDS, 1 barrier per K-step, register
// prefetch of next K-step; k2 stages the f16 activation tile with
// global_load_async_to_lds_b128 (ASYNCcnt) one step ahead.
// ---------------------------------------------------------------------------

typedef __attribute__((ext_vector_type(16))) _Float16 v16h;
typedef __attribute__((ext_vector_type(8)))  float    v8f;

#define HD      4096
#define MDIM    11008
#define TOK     512
#define NF4_BLK 64

__device__ const float NF4_TAB[16] = {
    -1.0f, -0.6961928009986877f, -0.5250730514526367f, -0.39491748809814453f,
    -0.28444138169288635f, -0.18477343022823334f, -0.09105003625154495f, 0.0f,
    0.07958029955625534f, 0.16093020141124725f, 0.24611230194568634f,
    0.33791524171829224f, 0.44070982933044434f, 0.5626170039176941f,
    0.7229568362236023f, 1.0f};

// 16x32 f16 fragment load (A layout == B layout), LDS row stride 32 halves.
//   lanes 0..15  : row = hl, K 0..7  in v0..3, K 16..23 in v4..7
//   lanes 16..31 : row = hl, K 8..15 in v0..3, K 24..31 in v4..7
__device__ __forceinline__ v16h ld_frag(const _Float16* __restrict__ tile, int row0) {
    const int lane = threadIdx.x & 31;
    const int hl   = lane & 15;
    const int hi   = lane >> 4;
    const _Float16* p = tile + (row0 + hl) * 32 + hi * 8;
    union { v16h v; uint4 q[2]; } u;
    u.q[0] = *(const uint4*)(p);
    u.q[1] = *(const uint4*)(p + 16);
    return u.v;
}

__device__ __forceinline__ v8f wmma_f16(v16h a, v16h b, v8f c) {
    return __builtin_amdgcn_wmma_f32_16x16x32_f16(
        false, a, false, b, (short)0, c, false, false);
}

// Direct global -> LDS async copy of 16 bytes (GLOBAL_LOAD_ASYNC_TO_LDS_B128,
// tracked by ASYNCcnt). lds_off = low 32 bits of the generic LDS pointer.
__device__ __forceinline__ void async_cp16(unsigned lds_off, const void* gaddr) {
    asm volatile("global_load_async_to_lds_b128 %0, %1, off"
                 :: "v"(lds_off), "v"(gaddr) : "memory");
}
__device__ __forceinline__ void wait_async0() {
    asm volatile("s_wait_asynccnt 0" ::: "memory");
}

// ---------------------------------------------------------------------------
// Kernel 1: h[t,m] = silu(x @ Wg^T) * (x @ Wu^T)  (f16 to workspace)
// Tile: 128 tokens x 64 m, BK=32, 8 waves; wave owns 32x32 (2x2 WMMA tiles)
// for BOTH gate and up. Double-buffered LDS, 1 barrier / K-step.
// ---------------------------------------------------------------------------
__global__ __launch_bounds__(256)
void nf4_gate_up_kernel(const float* __restrict__ x,
                        const int*   __restrict__ gcodes,
                        const float* __restrict__ gamax,
                        const int*   __restrict__ ucodes,
                        const float* __restrict__ uamax,
                        _Float16*    __restrict__ hbuf) {
    __shared__ __align__(16) _Float16 sA[2][128 * 32];
    __shared__ __align__(16) _Float16 sG[2][64 * 32];
    __shared__ __align__(16) _Float16 sU[2][64 * 32];
    __shared__ float s_nf4[16];

    const int tid = threadIdx.x;
    if (tid < 16) s_nf4[tid] = NF4_TAB[tid];
    __syncthreads();  // table ready before first dequant

    const int t0 = blockIdx.x * 128;
    const int n0 = blockIdx.y * 64;

    const int lane = tid & 31;
    const int wave = tid >> 5;
    const int wm   = (wave >> 1) * 32;
    const int wn   = (wave & 1) * 32;

    // register prefetch state
    float4 xr[4];
    int4   gcv[2], ucv[2];
    float  gam[2], uam[2];

    auto fetch = [&](int k0) {
        #pragma unroll
        for (int t = 0; t < 4; ++t) {
            const int idx = tid + t * 256;
            xr[t] = *(const float4*)(x + (size_t)(t0 + (idx >> 3)) * HD + k0 + (idx & 7) * 4);
        }
        #pragma unroll
        for (int t = 0; t < 2; ++t) {
            const int idx = tid + t * 256;
            const int row = idx >> 3, c4 = (idx & 7) * 4;
            const size_t roff = (size_t)(n0 + row) * HD + k0 + c4;
            const int    ab   = (n0 + row) * (HD / NF4_BLK) + (k0 >> 6);
            gcv[t] = *(const int4*)(gcodes + roff);
            gam[t] = gamax[ab];
            ucv[t] = *(const int4*)(ucodes + roff);
            uam[t] = uamax[ab];
        }
    };
    auto stage = [&](int buf) {
        #pragma unroll
        for (int t = 0; t < 4; ++t) {
            const int idx = tid + t * 256;
            _Float16* d = sA[buf] + (idx >> 3) * 32 + (idx & 7) * 4;
            d[0] = (_Float16)xr[t].x; d[1] = (_Float16)xr[t].y;
            d[2] = (_Float16)xr[t].z; d[3] = (_Float16)xr[t].w;
        }
        #pragma unroll
        for (int t = 0; t < 2; ++t) {
            const int idx = tid + t * 256;
            const int off = (idx >> 3) * 32 + (idx & 7) * 4;
            _Float16* dg = sG[buf] + off;
            dg[0] = (_Float16)(s_nf4[gcv[t].x & 15] * gam[t]);
            dg[1] = (_Float16)(s_nf4[gcv[t].y & 15] * gam[t]);
            dg[2] = (_Float16)(s_nf4[gcv[t].z & 15] * gam[t]);
            dg[3] = (_Float16)(s_nf4[gcv[t].w & 15] * gam[t]);
            _Float16* du = sU[buf] + off;
            du[0] = (_Float16)(s_nf4[ucv[t].x & 15] * uam[t]);
            du[1] = (_Float16)(s_nf4[ucv[t].y & 15] * uam[t]);
            du[2] = (_Float16)(s_nf4[ucv[t].z & 15] * uam[t]);
            du[3] = (_Float16)(s_nf4[ucv[t].w & 15] * uam[t]);
        }
    };

    v8f ag00 = {}, ag01 = {}, ag10 = {}, ag11 = {};
    v8f au00 = {}, au01 = {}, au10 = {}, au11 = {};

    fetch(0);
    int buf = 0;
    constexpr int NIT = HD / 32;  // 128
    for (int it = 0; it < NIT; ++it) {
        stage(buf);
        __syncthreads();
        if (it + 1 < NIT) fetch((it + 1) * 32);

        const v16h a0 = ld_frag(sA[buf], wm);
        const v16h a1 = ld_frag(sA[buf], wm + 16);
        const v16h g0 = ld_frag(sG[buf], wn);
        const v16h g1 = ld_frag(sG[buf], wn + 16);
        const v16h u0 = ld_frag(sU[buf], wn);
        const v16h u1 = ld_frag(sU[buf], wn + 16);

        ag00 = wmma_f16(a0, g0, ag00);  ag01 = wmma_f16(a0, g1, ag01);
        ag10 = wmma_f16(a1, g0, ag10);  ag11 = wmma_f16(a1, g1, ag11);
        au00 = wmma_f16(a0, u0, au00);  au01 = wmma_f16(a0, u1, au01);
        au10 = wmma_f16(a1, u0, au10);  au11 = wmma_f16(a1, u1, au11);
        buf ^= 1;
    }

    // epilogue: h = g * sigmoid(g) * u   (v_exp_f32 + v_rcp_f32, no IEEE div)
    // C/D layout: VGPR e, lanes 0..15 -> M=e, lanes 16..31 -> M=e+8; N=lane&15
    const int col  = lane & 15;
    const int rsel = (lane >> 4) * 8;

    auto store_h = [&](const v8f& g, const v8f& u, int r0, int c0) {
        #pragma unroll
        for (int e = 0; e < 8; ++e) {
            const float gv  = g[e];
            const float uv  = u[e];
            const float sig = __builtin_amdgcn_rcpf(1.0f + __expf(-gv));
            const float hv  = gv * sig * uv;
            const size_t r  = (size_t)(t0 + r0 + rsel + e);
            hbuf[r * MDIM + (n0 + c0 + col)] = (_Float16)hv;
        }
    };
    store_h(ag00, au00, wm,      wn);
    store_h(ag01, au01, wm,      wn + 16);
    store_h(ag10, au10, wm + 16, wn);
    store_h(ag11, au11, wm + 16, wn + 16);
}

// ---------------------------------------------------------------------------
// Kernel 2: out[t,hd] = h @ Wd^T, K = 11008. h tile staged global->LDS with
// async-to-LDS (issued one K-step ahead); Wd dequantized NF4 on the fly.
// ---------------------------------------------------------------------------
__global__ __launch_bounds__(256)
void nf4_down_kernel(const _Float16* __restrict__ hbuf,
                     const int*      __restrict__ dcodes,
                     const float*    __restrict__ damax,
                     float*          __restrict__ out) {
    __shared__ __align__(16) _Float16 sA[2][128 * 32];
    __shared__ __align__(16) _Float16 sW[2][64 * 32];
    __shared__ float s_nf4[16];

    const int tid = threadIdx.x;
    if (tid < 16) s_nf4[tid] = NF4_TAB[tid];
    __syncthreads();

    const int t0 = blockIdx.x * 128;
    const int n0 = blockIdx.y * 64;
    const int lane = tid & 31;
    const int wave = tid >> 5;
    const int wm   = (wave >> 1) * 32;
    const int wn   = (wave & 1) * 32;

    int4  dcv[2];
    float dam[2];

    // h tile: 128 rows x 32 halves = 512 x 16B chunks, 2 per thread
    auto issue_async = [&](int buf, int k0) {
        #pragma unroll
        for (int t = 0; t < 2; ++t) {
            const int idx = tid + t * 256;
            const int row = idx >> 2;
            const int c8  = (idx & 3) * 8;
            const _Float16* g = hbuf + (size_t)(t0 + row) * MDIM + k0 + c8;
            const unsigned  l = (unsigned)(uintptr_t)(const void*)(sA[buf] + row * 32 + c8);
            async_cp16(l, g);
        }
    };
    auto fetch_w = [&](int k0) {
        #pragma unroll
        for (int t = 0; t < 2; ++t) {
            const int idx = tid + t * 256;
            const int row = idx >> 3, c4 = (idx & 7) * 4;
            dcv[t] = *(const int4*)(dcodes + (size_t)(n0 + row) * MDIM + k0 + c4);
            dam[t] = damax[(n0 + row) * (MDIM / NF4_BLK) + (k0 >> 6)];
        }
    };
    auto stage_w = [&](int buf) {
        #pragma unroll
        for (int t = 0; t < 2; ++t) {
            const int idx = tid + t * 256;
            _Float16* d = sW[buf] + (idx >> 3) * 32 + (idx & 7) * 4;
            d[0] = (_Float16)(s_nf4[dcv[t].x & 15] * dam[t]);
            d[1] = (_Float16)(s_nf4[dcv[t].y & 15] * dam[t]);
            d[2] = (_Float16)(s_nf4[dcv[t].z & 15] * dam[t]);
            d[3] = (_Float16)(s_nf4[dcv[t].w & 15] * dam[t]);
        }
    };

    v8f a00 = {}, a01 = {}, a10 = {}, a11 = {};

    issue_async(0, 0);
    fetch_w(0);
    int buf = 0;
    constexpr int NIT = MDIM / 32;  // 344
    for (int it = 0; it < NIT; ++it) {
        stage_w(buf);
        wait_async0();        // this wave's async copies into sA[buf] landed
        __syncthreads();      // whole tile (all waves) visible
        if (it + 1 < NIT) {
            issue_async(buf ^ 1, (it + 1) * 32);  // prefetch next h tile
            fetch_w((it + 1) * 32);               // prefetch next codes
        }

        const v16h a0 = ld_frag(sA[buf], wm);
        const v16h a1 = ld_frag(sA[buf], wm + 16);
        const v16h w0 = ld_frag(sW[buf], wn);
        const v16h w1 = ld_frag(sW[buf], wn + 16);

        a00 = wmma_f16(a0, w0, a00);  a01 = wmma_f16(a0, w1, a01);
        a10 = wmma_f16(a1, w0, a10);  a11 = wmma_f16(a1, w1, a11);
        buf ^= 1;
    }

    const int col  = lane & 15;
    const int rsel = (lane >> 4) * 8;
    auto store_o = [&](const v8f& acc, int r0, int c0) {
        #pragma unroll
        for (int e = 0; e < 8; ++e) {
            const size_t r = (size_t)(t0 + r0 + rsel + e);
            out[r * HD + (n0 + c0 + col)] = acc[e];
        }
    };
    store_o(a00, wm,      wn);
    store_o(a01, wm,      wn + 16);
    store_o(a10, wm + 16, wn);
    store_o(a11, wm + 16, wn + 16);
}

// ---------------------------------------------------------------------------
extern "C" void kernel_launch(void* const* d_in, const int* in_sizes, int n_in,
                              void* d_out, int out_size, void* d_ws, size_t ws_size,
                              hipStream_t stream) {
    (void)in_sizes; (void)n_in; (void)out_size; (void)ws_size;

    const float* x      = (const float*)d_in[0];
    const int*   gcodes = (const int*)  d_in[1];
    const float* gamax  = (const float*)d_in[2];
    const int*   ucodes = (const int*)  d_in[3];
    const float* uamax  = (const float*)d_in[4];
    const int*   dcodes = (const int*)  d_in[5];
    const float* damax  = (const float*)d_in[6];
    float*       out    = (float*)d_out;
    _Float16*    hbuf   = (_Float16*)d_ws;   // 512*11008 f16 = ~11.3 MB

    dim3 g1(TOK / 128, MDIM / 64);   // (4, 172)
    nf4_gate_up_kernel<<<g1, 256, 0, stream>>>(x, gcodes, gamax, ucodes, uamax, hbuf);

    dim3 g2(TOK / 128, HD / 64);     // (4, 64)
    nf4_down_kernel<<<g2, 256, 0, stream>>>(hbuf, dcodes, damax, out);
}
